// SwinTransformerBlock_79302276153809
// MI455X (gfx1250) — compile-verified, matrix operand order checked by reading, not run on hardware
//
#include <hip/hip_runtime.h>
#include <hip/hip_bf16.h>

// ---------- CDNA5 WMMA types ----------
typedef _Float16 half8 __attribute__((ext_vector_type(8)));
typedef _Float16 v16h  __attribute__((ext_vector_type(16)));
typedef float    v8f   __attribute__((ext_vector_type(8)));

static constexpr int    C_    = 192;
static constexpr int    NH_   = 6;
static constexpr int    QKVN  = 576;   // 3*C
static constexpr int    HID   = 768;   // 4*C
static constexpr size_t TOKENS = 401408; // 8*224*224 = B*L, also nWin*49

__device__ __forceinline__ v16h cat8(half8 lo, half8 hi) {
  return __builtin_shufflevector(lo, hi, 0,1,2,3,4,5,6,7,8,9,10,11,12,13,14,15);
}
__device__ __forceinline__ v16h zfrag() {
  v16h z;
  #pragma unroll
  for (int i = 0; i < 16; ++i) z[i] = (_Float16)0.f;
  return z;
}
__device__ __forceinline__ v8f zacc() {
  v8f z;
  #pragma unroll
  for (int i = 0; i < 8; ++i) z[i] = 0.f;
  return z;
}
__device__ __forceinline__ v8f wmma16(v16h a, v16h b, v8f c) {
  // D = A(16x32 f16) * B(32x16 f16) + C(16x16 f32)
  return __builtin_amdgcn_wmma_f32_16x16x32_f16(false, a, false, b, (short)0, c, false, false);
}

// ---------- weight cast + transpose: W[K][N] f32 -> Wt[N][K] f16 ----------
__global__ __launch_bounds__(256)
void cast_transpose_k(const float* __restrict__ W, _Float16* __restrict__ Wt, int K, int N) {
  int t = blockIdx.x * 256 + threadIdx.x;
  if (t >= K * N) return;
  int k = t / N, n = t - k * N;
  Wt[(size_t)n * K + k] = (_Float16)W[(size_t)k * N + n];
}

// ---------- LayerNorm (+ optional shift/window-partition) -> f16 ----------
// WINDOWED=1: output index is windowed token order, source is rolled (-3,-3) natural order.
template<int WINDOWED>
__global__ __launch_bounds__(256)
void ln_cast_k(const float* __restrict__ x, const float* __restrict__ g,
               const float* __restrict__ bt, _Float16* __restrict__ out) {
  const int lane = threadIdx.x & 31;
  const int wid  = threadIdx.x >> 5;
  const unsigned widx = blockIdx.x * 8u + wid;   // output token index
  size_t src;
  if (WINDOWED) {
    unsigned win = widx / 49u, loc = widx % 49u;
    unsigned b = win >> 10, wi = win & 1023u;
    unsigned wh = wi >> 5, ww = wi & 31u;
    unsigned r7 = loc / 7u, c7 = loc % 7u;
    unsigned h = wh * 7u + r7, w = ww * 7u + c7;
    unsigned hs = h + 3u; if (hs >= 224u) hs -= 224u;   // roll(-3): shifted[h] = x[(h+3)%224]
    unsigned ws2 = w + 3u; if (ws2 >= 224u) ws2 -= 224u;
    src = (size_t)b * 50176u + (size_t)hs * 224u + ws2;
  } else {
    src = widx;
  }
  const float* xp = x + src * (size_t)C_;
  float v[6], s = 0.f, sq = 0.f;
  #pragma unroll
  for (int i = 0; i < 6; ++i) { v[i] = xp[lane + 32 * i]; s += v[i]; sq += v[i] * v[i]; }
  #pragma unroll
  for (int off = 1; off < 32; off <<= 1) { s += __shfl_xor(s, off, 32); sq += __shfl_xor(sq, off, 32); }
  const float mean = s * (1.f / 192.f);
  const float inv  = rsqrtf(sq * (1.f / 192.f) - mean * mean + 1e-5f);
  _Float16* op = out + (size_t)widx * C_;
  #pragma unroll
  for (int i = 0; i < 6; ++i) {
    int d = lane + 32 * i;
    op[d] = (_Float16)((v[i] - mean) * inv * g[d] + bt[d]);
  }
}

// ---------- WMMA GEMM: C[M,N] = A[M,K](f16) * Bt[N,K](f16)^T + bias ----------
// block tile 128(M) x 64(N); 8 waves, each 32x32 (2x2 wmma tiles); K step 32.
// EPI: 0 -> f16 store (qkv)
//      1 -> window-reverse + roll(+3,+3) scatter + residual(x) -> f32 (proj)
//      2 -> exact GELU -> f16 (fc1)
//      3 -> residual(xres) add -> f32 (fc2 / final output)
template<int EPI>
__global__ __launch_bounds__(256)
void gemm_wmma_k(const _Float16* __restrict__ A, const _Float16* __restrict__ Bt,
                 const float* __restrict__ bias, int K, int N,
                 _Float16* __restrict__ outH, float* __restrict__ outF,
                 const float* __restrict__ resid) {
  __shared__ alignas(16) _Float16 As[128][40];
  __shared__ alignas(16) _Float16 Bs[64][40];

  const int tid  = threadIdx.x;
  const int lane = tid & 31;
  const int wid  = tid >> 5;
  const int wm   = wid & 3;      // 4 waves along M
  const int wn   = wid >> 2;     // 2 waves along N
  const int c16  = lane & 15;
  const int hsel = lane >> 4;
  const int kb   = hsel * 8;

  const int    bn0 = blockIdx.x * 64;
  const size_t bm0 = (size_t)blockIdx.y * 128;

  v8f acc[2][2];
  #pragma unroll
  for (int i = 0; i < 2; ++i)
    #pragma unroll
    for (int j = 0; j < 2; ++j) acc[i][j] = zacc();

  for (int k0 = 0; k0 < K; k0 += 32) {
    // stage A tile: 128 rows x 32 halves (512 x 16B chunks)
    #pragma unroll
    for (int s = 0; s < 2; ++s) {
      int c = tid + s * 256;
      int row = c >> 2, cc = (c & 3) << 3;
      *(half8*)&As[row][cc] = *(const half8*)(A + (bm0 + row) * (size_t)K + k0 + cc);
    }
    // stage B tile: 64 rows x 32 halves (256 x 16B chunks)
    {
      int row = tid >> 2, cc = (tid & 3) << 3;
      *(half8*)&Bs[row][cc] = *(const half8*)(Bt + (size_t)(bn0 + row) * K + k0 + cc);
    }
    if (k0 + 32 < K) {   // global_prefetch_b8 for next K slice
      __builtin_prefetch(A + (bm0 + (tid >> 1)) * (size_t)K + k0 + 32, 0, 1);
      __builtin_prefetch(Bt + (size_t)(bn0 + (tid & 63)) * K + k0 + 32, 0, 1);
    }
    __syncthreads();

    v16h af[2], bf[2];
    #pragma unroll
    for (int mt = 0; mt < 2; ++mt) {
      int ar = wm * 32 + mt * 16 + c16;
      af[mt] = cat8(*(const half8*)&As[ar][kb], *(const half8*)&As[ar][16 + kb]);
    }
    #pragma unroll
    for (int nt = 0; nt < 2; ++nt) {
      int br = wn * 32 + nt * 16 + c16;
      bf[nt] = cat8(*(const half8*)&Bs[br][kb], *(const half8*)&Bs[br][16 + kb]);
    }
    #pragma unroll
    for (int mt = 0; mt < 2; ++mt)
      #pragma unroll
      for (int nt = 0; nt < 2; ++nt)
        acc[mt][nt] = wmma16(af[mt], bf[nt], acc[mt][nt]);
    __syncthreads();
  }

  // epilogue: C layout lane holds column gn; VGPR r holds row (hsel*8 + r)
  #pragma unroll
  for (int mt = 0; mt < 2; ++mt) {
    #pragma unroll
    for (int nt = 0; nt < 2; ++nt) {
      int gn = bn0 + wn * 32 + nt * 16 + c16;
      #pragma unroll
      for (int r = 0; r < 8; ++r) {
        size_t gm = bm0 + wm * 32 + mt * 16 + hsel * 8 + r;
        float v = acc[mt][nt][r] + bias[gn];
        if (EPI == 0) {
          outH[gm * (size_t)N + gn] = (_Float16)v;
        } else if (EPI == 2) {
          float gelu = 0.5f * v * (1.f + erff(v * 0.7071067811865475f));
          outH[gm * (size_t)N + gn] = (_Float16)gelu;
        } else if (EPI == 3) {
          outF[gm * (size_t)N + gn] = resid[gm * (size_t)N + gn] + v;
        } else { // EPI == 1: window reverse + unshift + residual
          unsigned t = (unsigned)gm;
          unsigned win = t / 49u, loc = t % 49u;
          unsigned b = win >> 10, wi = win & 1023u;
          unsigned wh = wi >> 5, ww = wi & 31u;
          unsigned r7 = loc / 7u, c7 = loc % 7u;
          unsigned h = wh * 7u + r7, w = ww * 7u + c7;
          unsigned h0 = h + 3u; if (h0 >= 224u) h0 -= 224u;   // roll(+3)
          unsigned w0 = w + 3u; if (w0 >= 224u) w0 -= 224u;
          size_t dst = (size_t)b * 50176u + (size_t)h0 * 224u + w0;
          outF[dst * (size_t)N + gn] = resid[dst * (size_t)N + gn] + v;
        }
      }
    }
  }
}

// ---------- windowed attention: one wave per (window, head) ----------
// S(64x64,pad) = q*k^T (16 WMMA) ; +scale +rpb +shift-mask ; softmax ; O = P*V (16 WMMA)
__global__ __launch_bounds__(128)
void attn_win_k(const _Float16* __restrict__ qkv, const float* __restrict__ rpb,
                _Float16* __restrict__ outp) {
  __shared__ alignas(16) _Float16 Pb[4][64][72];
  __shared__ alignas(16) _Float16 Vt[4][32][72];

  const int lane = threadIdx.x & 31;
  const int w    = threadIdx.x >> 5;
  const unsigned task = blockIdx.x * 4u + w;   // < 8192*6
  const unsigned win  = task / 6u;
  const int      head = (int)(task % 6u);
  const int c16 = lane & 15, hsel = lane >> 4, kb = hsel * 8;
  const unsigned wi = win & 1023u;
  const int wh = (int)(wi >> 5), ww = (int)(wi & 31u);

  const _Float16* base = qkv + (size_t)win * 49u * QKVN + head * 32;

  // A fragments for q (lane = row token, K = head dim chunks)
  v16h qf[4];
  #pragma unroll
  for (int i = 0; i < 4; ++i) {
    int tok = i * 16 + c16;
    if (tok < 49) {
      const _Float16* p = base + (size_t)tok * QKVN;
      qf[i] = cat8(*(const half8*)(p + kb), *(const half8*)(p + 16 + kb));
    } else qf[i] = zfrag();
  }

  // S = q * k^T
  v8f s[4][4];
  #pragma unroll
  for (int i = 0; i < 4; ++i)
    #pragma unroll
    for (int j = 0; j < 4; ++j) s[i][j] = zacc();
  #pragma unroll
  for (int j = 0; j < 4; ++j) {
    int tok = j * 16 + c16;
    v16h kf;
    if (tok < 49) {
      const _Float16* p = base + (size_t)tok * QKVN + 192;
      kf = cat8(*(const half8*)(p + kb), *(const half8*)(p + 16 + kb));
    } else kf = zfrag();
    #pragma unroll
    for (int i = 0; i < 4; ++i) s[i][j] = wmma16(qf[i], kf, s[i][j]);
  }

  // stage V^T into LDS (zero padded columns)
  for (int t = lane; t < 32 * 72; t += 32) ((_Float16*)Vt[w])[t] = (_Float16)0.f;
  for (int tok = lane; tok < 49; tok += 32) {
    const _Float16* p = base + (size_t)tok * QKVN + 384;
    #pragma unroll
    for (int d = 0; d < 32; ++d) Vt[w][d][tok] = p[d];
  }

  // scale + rpb + shift mask + row softmax -> P (f16 in LDS)
  const float scale = 0.17677669529663687f;  // 32^-0.5
  #pragma unroll
  for (int i = 0; i < 4; ++i) {
    #pragma unroll
    for (int r = 0; r < 8; ++r) {
      int m = i * 16 + hsel * 8 + r;
      int mh = m / 7, mw = m % 7;
      int hm = wh * 7 + mh, wm2 = ww * 7 + mw;
      int idm = (hm < 217 ? 0 : (hm < 221 ? 1 : 2)) * 3 + (wm2 < 217 ? 0 : (wm2 < 221 ? 1 : 2));
      float vj[4];
      #pragma unroll
      for (int j = 0; j < 4; ++j) {
        int n = j * 16 + c16;
        float v;
        if (m < 49 && n < 49) {
          int nh = n / 7, nw = n % 7;
          int hn = wh * 7 + nh, wn2 = ww * 7 + nw;
          int idn = (hn < 217 ? 0 : (hn < 221 ? 1 : 2)) * 3 + (wn2 < 217 ? 0 : (wn2 < 221 ? 1 : 2));
          int ridx = (mh - nh + 6) * 13 + (mw - nw + 6);
          v = s[i][j][r] * scale + rpb[ridx * NH_ + head] + (idm != idn ? -100.f : 0.f);
        } else v = -1e30f;
        vj[j] = v;
      }
      float mx = fmaxf(fmaxf(vj[0], vj[1]), fmaxf(vj[2], vj[3]));
      #pragma unroll
      for (int off = 1; off < 16; off <<= 1) mx = fmaxf(mx, __shfl_xor(mx, off, 32));
      float sum = 0.f;
      #pragma unroll
      for (int j = 0; j < 4; ++j) { vj[j] = __expf(vj[j] - mx); sum += vj[j]; }
      #pragma unroll
      for (int off = 1; off < 16; off <<= 1) sum += __shfl_xor(sum, off, 32);
      float inv = 1.f / sum;
      #pragma unroll
      for (int j = 0; j < 4; ++j) Pb[w][m][j * 16 + c16] = (_Float16)(vj[j] * inv);
    }
  }

  // O = P * V
  v8f o[4][2];
  #pragma unroll
  for (int i = 0; i < 4; ++i) { o[i][0] = zacc(); o[i][1] = zacc(); }
  #pragma unroll
  for (int ko = 0; ko < 2; ++ko) {
    v16h vb[2];
    #pragma unroll
    for (int nt = 0; nt < 2; ++nt) {
      int d = nt * 16 + c16;
      vb[nt] = cat8(*(const half8*)&Vt[w][d][ko * 32 + kb],
                    *(const half8*)&Vt[w][d][ko * 32 + 16 + kb]);
    }
    #pragma unroll
    for (int i = 0; i < 4; ++i) {
      int row = i * 16 + c16;
      v16h pa = cat8(*(const half8*)&Pb[w][row][ko * 32 + kb],
                     *(const half8*)&Pb[w][row][ko * 32 + 16 + kb]);
      o[i][0] = wmma16(pa, vb[0], o[i][0]);
      o[i][1] = wmma16(pa, vb[1], o[i][1]);
    }
  }

  // store O -> attnout[token, head*32 + d] (f16)
  #pragma unroll
  for (int i = 0; i < 4; ++i)
    #pragma unroll
    for (int nt = 0; nt < 2; ++nt)
      #pragma unroll
      for (int r = 0; r < 8; ++r) {
        int m = i * 16 + hsel * 8 + r;
        if (m < 49) {
          size_t t = (size_t)win * 49u + m;
          outp[t * (size_t)C_ + head * 32 + nt * 16 + c16] = (_Float16)o[i][nt][r];
        }
      }
}

// ---------------------------------------------------------------------------
extern "C" void kernel_launch(void* const* d_in, const int* in_sizes, int n_in,
                              void* d_out, int out_size, void* d_ws, size_t ws_size,
                              hipStream_t stream) {
  const float* x      = (const float*)d_in[0];
  const float* n1g    = (const float*)d_in[1];
  const float* n1b    = (const float*)d_in[2];
  const float* qkv_w  = (const float*)d_in[3];
  const float* qkv_b  = (const float*)d_in[4];
  const float* rpb    = (const float*)d_in[5];
  const float* proj_w = (const float*)d_in[6];
  const float* proj_b = (const float*)d_in[7];
  const float* n2g    = (const float*)d_in[8];
  const float* n2b    = (const float*)d_in[9];
  const float* fc1_w  = (const float*)d_in[10];
  const float* fc1_b  = (const float*)d_in[11];
  const float* fc2_w  = (const float*)d_in[12];
  const float* fc2_b  = (const float*)d_in[13];
  float* out = (float*)d_out;

  char* ws = (char*)d_ws;
  size_t off = 0;
  auto carve = [&](size_t bytes) { void* p = ws + off; off += (bytes + 255) & ~(size_t)255; return p; };
  _Float16* xw    = (_Float16*)carve(TOKENS * C_   * 2);  // LN1/windowed x ; reused as attnout
  _Float16* qkvb  = (_Float16*)carve(TOKENS * QKVN * 2);  // qkv f16 ; reused as h2
  float*    xres  = (float*)   carve(TOKENS * C_   * 4);  // shortcut + attn (f32)
  _Float16* h3    = (_Float16*)carve(TOKENS * HID  * 2);  // gelu(fc1) f16
  _Float16* qkvwT = (_Float16*)carve((size_t)C_  * QKVN * 2);
  _Float16* projT = (_Float16*)carve((size_t)C_  * C_   * 2);
  _Float16* fc1T  = (_Float16*)carve((size_t)C_  * HID  * 2);
  _Float16* fc2T  = (_Float16*)carve((size_t)HID * C_   * 2);
  _Float16* attno = xw;    // reuse
  _Float16* h2    = qkvb;  // reuse

  // weight prep (f32 -> f16, transposed to [N][K])
  cast_transpose_k<<<(C_ * QKVN + 255) / 256, 256, 0, stream>>>(qkv_w, qkvwT, C_, QKVN);
  cast_transpose_k<<<(C_ * C_   + 255) / 256, 256, 0, stream>>>(proj_w, projT, C_, C_);
  cast_transpose_k<<<(C_ * HID  + 255) / 256, 256, 0, stream>>>(fc1_w, fc1T, C_, HID);
  cast_transpose_k<<<(HID * C_  + 255) / 256, 256, 0, stream>>>(fc2_w, fc2T, HID, C_);

  const unsigned mBlocks = (unsigned)(TOKENS / 128);  // 3136
  const unsigned lnBlocks = (unsigned)(TOKENS / 8);   // 50176

  // 1) LN1 + roll(-3,-3) + window partition -> f16
  ln_cast_k<1><<<lnBlocks, 256, 0, stream>>>(x, n1g, n1b, xw);
  // 2) qkv = xw @ qkv_w + b -> f16
  gemm_wmma_k<0><<<dim3(QKVN / 64, mBlocks), 256, 0, stream>>>(xw, qkvwT, qkv_b, C_, QKVN,
                                                               qkvb, nullptr, nullptr);
  // 3) windowed attention (8192 windows x 6 heads, 1 wave each)
  attn_win_k<<<(8192u * 6u) / 4u, 128, 0, stream>>>(qkvb, rpb, attno);
  // 4) proj + window reverse + roll(+3,+3) + residual -> xres (f32)
  gemm_wmma_k<1><<<dim3(C_ / 64, mBlocks), 256, 0, stream>>>(attno, projT, proj_b, C_, C_,
                                                             nullptr, xres, x);
  // 5) LN2 -> f16
  ln_cast_k<0><<<lnBlocks, 256, 0, stream>>>(xres, n2g, n2b, h2);
  // 6) fc1 + GELU -> f16
  gemm_wmma_k<2><<<dim3(HID / 64, mBlocks), 256, 0, stream>>>(h2, fc1T, fc1_b, C_, HID,
                                                              h3, nullptr, nullptr);
  // 7) fc2 + residual -> final f32 output
  gemm_wmma_k<3><<<dim3(C_ / 64, mBlocks), 256, 0, stream>>>(h3, fc2T, fc2_b, HID, C_,
                                                             nullptr, out, xres);
  (void)in_sizes; (void)n_in; (void)out_size; (void)ws_size;
}
